// TGACN_63307817943092
// MI455X (gfx1250) — matrix operand order repaired and checked by reading, not compiled, not run on hardware
//
#include <hip/hip_runtime.h>

#define Bz 32
#define Cc 64
#define Nn 512
#define Tt 24
#define Oo 192
#define O3 64
#define NT (Nn * Tt)                      // 12288
#define SH ((size_t)Bz * Cc * Nn * Tt)    // 25,165,824 elements

typedef __attribute__((ext_vector_type(16))) __bf16 v16bf;
typedef __attribute__((ext_vector_type(8)))  float  v8f;

// Two ds_load_b128 -> one 16x32 (A) or 32x16 (B) bf16 WMMA fragment.
__device__ inline v16bf load_frag(const int4* p) {
  union { int4 q[2]; v16bf v; } u;
  u.q[0] = p[0];
  u.q[1] = p[1];
  return u.v;
}

// ---------------------------------------------------------------------------
// Weight packing to bf16 (conv taps: tap0 = w[...,1] (current), tap1 = w[...,0]
// (sample at t-d)).
// ---------------------------------------------------------------------------
__global__ void k_pack(const float* __restrict__ gacn_W, const float* __restrict__ res_w,
                       const float* __restrict__ w1, const float* __restrict__ w2,
                       __bf16* __restrict__ gW, __bf16* __restrict__ rW,
                       __bf16* __restrict__ w1p, __bf16* __restrict__ w2p) {
  int i = blockIdx.x * blockDim.x + threadIdx.x;
  if (i < O3 * Cc) gW[i] = (__bf16)gacn_W[i];
  if (i < Oo * Cc) rW[i] = (__bf16)res_w[i];          // res_w[:,:,0,0] is flat [O][C]
  if (i < O3 * 2 * Cc) {
    int o = i >> 7, k = i & 127, c = k & 63, tap = k >> 6;
    int wi = (o * Cc + c) * 2 + (tap ? 0 : 1);
    w1p[i] = (__bf16)w1[wi];
    w2p[i] = (__bf16)w2[wi];
  }
}

// ---------------------------------------------------------------------------
// LayerNorm over T; emit h (bf16, [b,c,n,t]), hT (bf16, [b,n,c,t]), x (bf16).
// ---------------------------------------------------------------------------
__global__ __launch_bounds__(256) void k_ln(const float* __restrict__ x,
                                            const float* __restrict__ lnw,
                                            const float* __restrict__ lnb,
                                            __bf16* __restrict__ h,
                                            __bf16* __restrict__ hT,
                                            __bf16* __restrict__ xb) {
  size_t row = (size_t)blockIdx.x * blockDim.x + threadIdx.x;  // over B*C*N
  int n = (int)(row % Nn);
  int c = (int)((row / Nn) % Cc);
  int b = (int)(row / ((size_t)Cc * Nn));
  const float* xr = x + row * Tt;
  float mu = 0.f;
#pragma unroll
  for (int t = 0; t < Tt; ++t) mu += xr[t];
  mu *= (1.f / Tt);
  float var = 0.f;
#pragma unroll
  for (int t = 0; t < Tt; ++t) { float d = xr[t] - mu; var += d * d; }
  var *= (1.f / Tt);
  float rstd = rsqrtf(var + 1e-5f);
  __bf16* hr  = h  + row * Tt;
  __bf16* hTr = hT + (((size_t)b * Nn + n) * Cc + c) * Tt;
  __bf16* xbr = xb + row * Tt;
#pragma unroll
  for (int t = 0; t < Tt; ++t) {
    float hv = (xr[t] - mu) * rstd * lnw[t] + lnb[t];
    hr[t]  = (__bf16)hv;
    hTr[t] = (__bf16)hv;
    xbr[t] = (__bf16)xr[t];
  }
}

// kc[b,c,t] = sum_n alpha[n] * h[b,c,n,t]
__global__ void k_kc(const __bf16* __restrict__ h, const float* __restrict__ alpha,
                     float* __restrict__ kc) {
  int idx = blockIdx.x * blockDim.x + threadIdx.x;  // B*C*T
  int t = idx % Tt, c = (idx / Tt) % Cc, b = idx / (Tt * Cc);
  const __bf16* hp = h + (((size_t)b * Cc + c) * Nn) * Tt + t;
  float s = 0.f;
  for (int n = 0; n < Nn; ++n) s += alpha[n] * (float)hp[(size_t)n * Tt];
  kc[idx] = s;
}

// kg[b,n,t] = sum_c ga[c]*h ; kt[b,t,n] = sum_c ta[c]*h
__global__ void k_kgkt(const __bf16* __restrict__ h, const float* __restrict__ ga,
                       const float* __restrict__ ta, float* __restrict__ kg,
                       float* __restrict__ kt) {
  int idx = blockIdx.x * blockDim.x + threadIdx.x;  // B*N*T
  int t = idx % Tt, n = (idx / Tt) % Nn, b = idx / (Tt * Nn);
  const __bf16* hp = h + (((size_t)b * Cc) * Nn + n) * Tt + t;
  float sg = 0.f, st = 0.f;
  for (int c = 0; c < Cc; ++c) {
    float hv = (float)hp[(size_t)c * Nn * Tt];
    sg += ga[c] * hv;
    st += ta[c] * hv;
  }
  kg[idx] = sg;
  kt[((size_t)b * Tt + t) * Nn + n] = st;
}

// kout[r,s] = sum_t kin[r,t] * W[t,s]   (T x T projection)
__global__ void k_kW(const float* __restrict__ kin, const float* __restrict__ W,
                     float* __restrict__ kout, int rows) {
  int idx = blockIdx.x * blockDim.x + threadIdx.x;
  if (idx >= rows * Tt) return;
  int s = idx % Tt, r = idx / Tt;
  const float* kr = kin + (size_t)r * Tt;
  float acc = 0.f;
#pragma unroll
  for (int t = 0; t < Tt; ++t) acc += kr[t] * W[t * Tt + s];
  kout[idx] = acc;
}

// att_c: one 64-thread block per (b,c); softmax over d (=64)
__global__ __launch_bounds__(64) void k_att_c(const float* __restrict__ kcW,
                                              const float* __restrict__ kc,
                                              __bf16* __restrict__ attC) {
  __shared__ float red[64];
  int bc = blockIdx.x, b = bc / Cc, d = threadIdx.x;
  const float* qr = kcW + (size_t)bc * Tt;
  const float* kr = kc + ((size_t)b * Cc + d) * Tt;
  float s = 0.f;
#pragma unroll
  for (int t = 0; t < Tt; ++t) s += qr[t] * kr[t];
  red[d] = s; __syncthreads();
  for (int st = 32; st > 0; st >>= 1) { if (d < st) red[d] = fmaxf(red[d], red[d + st]); __syncthreads(); }
  float mx = red[0]; __syncthreads();
  float e = __expf(s - mx);
  red[d] = e; __syncthreads();
  for (int st = 32; st > 0; st >>= 1) { if (d < st) red[d] += red[d + st]; __syncthreads(); }
  attC[(size_t)bc * Cc + d] = (__bf16)(e / red[0]);
}

// att_g: one 256-thread block per (b,n); softmax over m (=512), then *Adj
__global__ __launch_bounds__(256) void k_att_g(const float* __restrict__ kgW,
                                               const float* __restrict__ kg,
                                               const float* __restrict__ Adj,
                                               __bf16* __restrict__ attG) {
  __shared__ float lg[Nn];
  __shared__ float red[256];
  int bn = blockIdx.x, b = bn / Nn, n = bn % Nn, tid = threadIdx.x;
  float qv[Tt];
  const float* qr = kgW + (size_t)bn * Tt;
#pragma unroll
  for (int t = 0; t < Tt; ++t) qv[t] = qr[t];
  for (int m = tid; m < Nn; m += 256) {
    const float* kr = kg + ((size_t)b * Nn + m) * Tt;
    float s = 0.f;
#pragma unroll
    for (int t = 0; t < Tt; ++t) s += qv[t] * kr[t];
    lg[m] = s;
  }
  __syncthreads();
  red[tid] = fmaxf(lg[tid], lg[tid + 256]); __syncthreads();
  for (int st = 128; st > 0; st >>= 1) { if (tid < st) red[tid] = fmaxf(red[tid], red[tid + st]); __syncthreads(); }
  float mx = red[0]; __syncthreads();
  float e0 = __expf(lg[tid] - mx), e1 = __expf(lg[tid + 256] - mx);
  red[tid] = e0 + e1; __syncthreads();
  for (int st = 128; st > 0; st >>= 1) { if (tid < st) red[tid] += red[tid + st]; __syncthreads(); }
  float inv = 1.f / red[0];
  const float* arow = Adj + (size_t)n * Nn;
  __bf16* orow = attG + (size_t)bn * Nn;
  orow[tid]       = (__bf16)(e0 * inv * arow[tid]);
  orow[tid + 256] = (__bf16)(e1 * inv * arow[tid + 256]);
}

// att_t (rank-10): one block per b; emits transposed bf16 attTT[b][i][t]
__global__ __launch_bounds__(256) void k_att_t(const float* __restrict__ kt,
                                               const float* __restrict__ W1,
                                               const float* __restrict__ W2,
                                               __bf16* __restrict__ attTT) {
  __shared__ float skt[Tt][Nn];
  __shared__ float q[Tt][10];
  __shared__ float p[Tt][10];
  int b = blockIdx.x, tid = threadIdx.x;
  for (int i = tid; i < Tt * Nn; i += 256) skt[i / Nn][i % Nn] = kt[(size_t)b * Tt * Nn + i];
  __syncthreads();
  for (int task = tid; task < 2 * Tt * 10; task += 256) {
    int which = task / (Tt * 10), r = task % (Tt * 10), t = r / 10, k = r % 10;
    const float* W = which ? W2 : W1;
    float s = 0.f;
    for (int n = 0; n < Nn; ++n) s += skt[t][n] * W[k * Nn + n];
    if (which) p[t][k] = s; else q[t][k] = s;
  }
  __syncthreads();
  if (tid < Tt) {
    int t = tid;
    float lg[Tt];
    float mx = -3.4e38f;
#pragma unroll
    for (int s = 0; s < Tt; ++s) {
      float v = 0.f;
#pragma unroll
      for (int k = 0; k < 10; ++k) v += q[t][k] * p[s][k];
      lg[s] = v; mx = fmaxf(mx, v);
    }
    float sum = 0.f;
#pragma unroll
    for (int s = 0; s < Tt; ++s) { float e = __expf(lg[s] - mx); lg[s] = e; sum += e; }
    float inv = 1.f / sum;
#pragma unroll
    for (int s = 0; s < Tt; ++s)
      attTT[(size_t)b * Tt * Tt + s * Tt + t] = (__bf16)(lg[s] * inv);
  }
}

// ---------------------------------------------------------------------------
// Batched bf16 WMMA GEMM: Out[b] = A[b](MxK) @ B[b](KxN) (+bias[M]).
// 64x64 tile / block, 8 waves, wave = one 16x32 D strip (2 WMMA accumulators).
// LDS tiles are stored directly in WMMA *fragment* layout (ISA 7.12.2):
//   A frag: lane L holds row (mt*16 + L%16), K-elems e=0..7 -> K=(L>=16?8:0)+e,
//           e=8..15 -> K=16+(L>=16?8:0)+(e-8)   (contiguous 32B per lane)
//   B frag: lane L holds col L%16, elems e -> same K order as A frag
// so each fragment is two ds_load_b128. Double-buffered K loop, one barrier
// per step, next tile's global_load_b128s issued before the WMMAs.
// GUARDED=false requires M%64==0, K%32==0, N%64==0 (all big GEMMs here).
// ---------------------------------------------------------------------------
template <bool GUARDED>
__global__ __launch_bounds__(256) void k_gemm(const __bf16* __restrict__ A,
                                              const __bf16* __restrict__ Bm,
                                              void* __restrict__ OutP,
                                              const float* __restrict__ bias,
                                              int M, int K, int Ncols,
                                              long long sA, long long sB, long long sO,
                                              int out_f32) {
  __shared__ int4 lA[2][4][32][2];   // [buf][mtile][lane][16B half]
  __shared__ int4 lB[2][4][32][2];   // [buf][ntile][lane][16B half]
  int b = blockIdx.z;
  const __bf16* Ab = A + (size_t)b * sA;
  const __bf16* Bb = Bm + (size_t)b * sB;
  int tM = blockIdx.y * 64, tN = blockIdx.x * 64;
  int tid = threadIdx.x, lane = tid & 31, wave = tid >> 5;
  int mt = wave >> 1, nt0 = (wave & 1) * 2;

  // A staging: thread loads 8 contiguous K-elems of row ar (one fragment half).
  int ar = tid >> 2, aks = (tid & 3) * 8;
  int amt = ar >> 4;
  int alane = ((aks & 8) ? 16 : 0) + (ar & 15);
  int ahalf = (aks & 16) ? 1 : 0;
  // B staging: thread loads row bkk, cols bcc0..bcc0+7 (scatter over 8 lanes).
  int bkk = tid >> 3, bcc0 = (tid & 7) * 8;
  int blb = (bkk & 8) ? 16 : 0, be = (bkk & 7) + ((bkk & 16) ? 8 : 0);

  union AV { __bf16 e[8]; int4 q; };
  AV va, vb;

  auto load_step = [&](int k0) {
    if (!GUARDED) {
      va.q = *(const int4*)(Ab + (size_t)(tM + ar) * K + k0 + aks);
      vb.q = *(const int4*)(Bb + (size_t)(k0 + bkk) * Ncols + tN + bcc0);
    } else {
      int gm = tM + ar;
#pragma unroll
      for (int i = 0; i < 8; ++i) {
        int gk = k0 + aks + i;
        va.e[i] = (gm < M && gk < K) ? Ab[(size_t)gm * K + gk] : (__bf16)0.f;
      }
      int gk = k0 + bkk;
#pragma unroll
      for (int i = 0; i < 8; ++i) {
        int gc = tN + bcc0 + i;
        vb.e[i] = (gk < K && gc < Ncols) ? Bb[(size_t)gk * Ncols + gc] : (__bf16)0.f;
      }
    }
  };
  auto store_step = [&](int buf) {
    lA[buf][amt][alane][ahalf] = va.q;
#pragma unroll
    for (int i = 0; i < 8; ++i) {
      int cc = bcc0 + i;
      ((__bf16*)&lB[buf][cc >> 4][blb + (cc & 15)][0])[be] = vb.e[i];
    }
  };

  v8f acc0 = {}; v8f acc1 = {};
  int nsteps = (K + 31) >> 5;
  load_step(0);
  store_step(0);
  for (int s = 0; s < nsteps; ++s) {
    __syncthreads();
    int cur = s & 1;
    bool more = (s + 1) < nsteps;
    if (more) load_step((s + 1) * 32);
    v16bf af  = load_frag(&lA[cur][mt][lane][0]);
    v16bf bf0 = load_frag(&lB[cur][nt0][lane][0]);
    v16bf bf1 = load_frag(&lB[cur][nt0 + 1][lane][0]);
    acc0 = __builtin_amdgcn_wmma_f32_16x16x32_bf16(false, af, false, bf0, (short)0, acc0, false, false);
    acc1 = __builtin_amdgcn_wmma_f32_16x16x32_bf16(false, af, false, bf1, (short)0, acc1, false, false);
    if (more) store_step(1 - cur);
  }

  int rbase = mt * 16 + ((lane >> 4) << 3);
  int gc0 = tN + nt0 * 16 + (lane & 15);
#pragma unroll
  for (int r = 0; r < 8; ++r) {
    int gm = tM + rbase + r;
    if (GUARDED && gm >= M) continue;
    float v0 = acc0[r], v1 = acc1[r];
    if (bias) { float bv = bias[gm]; v0 += bv; v1 += bv; }
    if (out_f32) {
      float* Ob = (float*)OutP + (size_t)b * sO;
      if (!GUARDED || gc0 < Ncols)      Ob[(size_t)gm * Ncols + gc0]      = v0;
      if (!GUARDED || gc0 + 16 < Ncols) Ob[(size_t)gm * Ncols + gc0 + 16] = v1;
    } else {
      __bf16* Ob = (__bf16*)OutP + (size_t)b * sO;
      if (!GUARDED || gc0 < Ncols)      Ob[(size_t)gm * Ncols + gc0]      = (__bf16)v0;
      if (!GUARDED || gc0 + 16 < Ncols) Ob[(size_t)gm * Ncols + gc0 + 16] = (__bf16)v1;
    }
  }
}

// ---------------------------------------------------------------------------
// Causal dilated conv (width 2) as implicit GEMM: K=128 (2 taps x 64 chans).
// Whole K staged once in fragment layout; 4 unrolled WMMA steps, no K-loop
// barriers. Tap-1 rows are the input shifted by -dil in time (zero for t<dil).
// ---------------------------------------------------------------------------
__global__ __launch_bounds__(256) void k_conv(const __bf16* __restrict__ Wp,
                                              const __bf16* __restrict__ In,
                                              __bf16* __restrict__ Out,
                                              const float* __restrict__ bias,
                                              int dil) {
  __shared__ int4 lA[4][4][32][2];   // [kstep][mtile][lane][half]
  __shared__ int4 lB[4][4][32][2];
  int b = blockIdx.z;
  int tN = blockIdx.x * 64;
  int tid = threadIdx.x, lane = tid & 31, wave = tid >> 5;
  int mt = wave >> 1, nt0 = (wave & 1) * 2;
  const __bf16* Inb = In + (size_t)b * Cc * NT;
  union AV { __bf16 e[8]; int4 q; };

  // A: 64x128 weights -> fragment layout (4 groups of 8 per thread).
#pragma unroll
  for (int j = 0; j < 4; ++j) {
    int g = tid * 4 + j;
    int r = g >> 4, kks = (g & 15) * 8;
    int ks = kks >> 5, kk = kks & 31;
    AV v;
    v.q = *(const int4*)(Wp + r * 128 + kks);
    lA[ks][r >> 4][((kk & 8) ? 16 : 0) + (r & 15)][(kk & 16) ? 1 : 0] = v.q;
  }
  // B: logical 128 x 64 tile (tap0 rows vector-loaded, tap1 rows shifted).
#pragma unroll
  for (int j = 0; j < 4; ++j) {
    int g = tid * 4 + j;
    int kk = g >> 3, cc0 = (g & 7) * 8;
    int gc0 = tN + cc0;
    int tap = kk >> 6, c = kk & 63;
    int ks = kk >> 5, kk32 = kk & 31;
    int blb = (kk32 & 8) ? 16 : 0, be = (kk32 & 7) + ((kk32 & 16) ? 8 : 0);
    AV v;
    if (tap == 0) {
      v.q = *(const int4*)(Inb + (size_t)c * NT + gc0);
    } else {
#pragma unroll
      for (int i = 0; i < 8; ++i) {
        int gc = gc0 + i;
        int t = gc % Tt;
        __bf16 xv = Inb[(size_t)c * NT + (t >= dil ? gc - dil : gc)];
        v.e[i] = (t >= dil) ? xv : (__bf16)0.f;
      }
    }
#pragma unroll
    for (int i = 0; i < 8; ++i) {
      int cc = cc0 + i;
      ((__bf16*)&lB[ks][cc >> 4][blb + (cc & 15)][0])[be] = v.e[i];
    }
  }
  __syncthreads();

  v8f acc0 = {}; v8f acc1 = {};
#pragma unroll
  for (int ks = 0; ks < 4; ++ks) {
    v16bf af  = load_frag(&lA[ks][mt][lane][0]);
    v16bf bf0 = load_frag(&lB[ks][nt0][lane][0]);
    v16bf bf1 = load_frag(&lB[ks][nt0 + 1][lane][0]);
    acc0 = __builtin_amdgcn_wmma_f32_16x16x32_bf16(false, af, false, bf0, (short)0, acc0, false, false);
    acc1 = __builtin_amdgcn_wmma_f32_16x16x32_bf16(false, af, false, bf1, (short)0, acc1, false, false);
  }
  int rbase = mt * 16 + ((lane >> 4) << 3);
  int gc0 = tN + nt0 * 16 + (lane & 15);
  __bf16* Ob = Out + (size_t)b * Cc * NT;
#pragma unroll
  for (int r = 0; r < 8; ++r) {
    int gm = rbase + r;
    float bv = bias[gm];
    Ob[(size_t)gm * NT + gc0]      = (__bf16)(acc0[r] + bv);
    Ob[(size_t)gm * NT + gc0 + 16] = (__bf16)(acc1[r] + bv);
  }
}

// hWT[b,n,o,t] = hW1[b,o,n,t]
__global__ void k_transpose(const __bf16* __restrict__ src, __bf16* __restrict__ dst) {
  size_t idx = (size_t)blockIdx.x * 256 + threadIdx.x;  // over SH
  int t = (int)(idx % Tt);
  size_t r = idx / Tt;
  int n = (int)(r % Nn);
  size_t r2 = r / Nn;
  int o = (int)(r2 % O3);
  int b = (int)(r2 / O3);
  dst[(((size_t)b * Nn + n) * O3 + o) * Tt + t] = src[idx];
}

// out = relu(res + concat(c_out, g_out, t_out))
__global__ void k_final(float* __restrict__ out, const __bf16* __restrict__ cO,
                        const __bf16* __restrict__ gO, const __bf16* __restrict__ tO) {
  size_t idx = (size_t)blockIdx.x * 256 + threadIdx.x;  // over B*O*N*T
  int t = (int)(idx % Tt);
  size_t r = idx / Tt;
  int n = (int)(r % Nn);
  size_t r2 = r / Nn;
  int o = (int)(r2 % Oo);
  int b = (int)(r2 / Oo);
  float br;
  if (o < O3)            br = (float)cO[(((size_t)b * Cc + o) * Nn + n) * Tt + t];
  else if (o < 2 * O3)   br = (float)gO[(((size_t)b * Nn + n) * O3 + (o - O3)) * Tt + t];
  else                   br = (float)tO[(((size_t)b * O3 + (o - 2 * O3)) * Nn + n) * Tt + t];
  float v = out[idx] + br;
  out[idx] = v > 0.f ? v : 0.f;
}

// ---------------------------------------------------------------------------
extern "C" void kernel_launch(void* const* d_in, const int* in_sizes, int n_in,
                              void* d_out, int out_size, void* d_ws, size_t ws_size,
                              hipStream_t stream) {
  (void)in_sizes; (void)n_in; (void)out_size; (void)ws_size;
  const float* x       = (const float*)d_in[0];
  const float* Adj     = (const float*)d_in[1];
  const float* ln_w    = (const float*)d_in[2];
  const float* ln_b    = (const float*)d_in[3];
  const float* catt_W  = (const float*)d_in[4];
  const float* catt_al = (const float*)d_in[5];
  const float* gatt_W  = (const float*)d_in[6];
  const float* gatt_al = (const float*)d_in[7];
  const float* gacn_W  = (const float*)d_in[8];
  const float* tatt_W1 = (const float*)d_in[9];
  const float* tatt_W2 = (const float*)d_in[10];
  const float* tatt_al = (const float*)d_in[11];
  const float* tcn1_w  = (const float*)d_in[12];
  const float* tcn1_b  = (const float*)d_in[13];
  const float* tcn2_w  = (const float*)d_in[14];
  const float* tcn2_b  = (const float*)d_in[15];
  const float* res_w   = (const float*)d_in[16];
  const float* res_b   = (const float*)d_in[17];
  float* out = (float*)d_out;

  size_t off = 0;
  auto alloc = [&](size_t bytes) -> void* {
    void* p = (char*)d_ws + off;
    off += (bytes + 255) & ~(size_t)255;
    return p;
  };
  __bf16* h     = (__bf16*)alloc(SH * 2);
  __bf16* hT    = (__bf16*)alloc(SH * 2);
  __bf16* xb    = (__bf16*)alloc(SH * 2);
  __bf16* cOut  = (__bf16*)alloc(SH * 2);
  __bf16* hW1   = (__bf16*)alloc(SH * 2);   // reused as g_out after transpose
  __bf16* hWT   = (__bf16*)alloc(SH * 2);
  __bf16* tTmp  = (__bf16*)alloc(SH * 2);   // reused as t_out after conv1
  __bf16* y1    = (__bf16*)alloc(SH * 2);
  __bf16* attG  = (__bf16*)alloc((size_t)Bz * Nn * Nn * 2);
  __bf16* attC  = (__bf16*)alloc((size_t)Bz * Cc * Cc * 2);
  __bf16* attTT = (__bf16*)alloc((size_t)Bz * Tt * Tt * 2);
  float*  kc    = (float*)alloc((size_t)Bz * Cc * Tt * 4);
  float*  kcW   = (float*)alloc((size_t)Bz * Cc * Tt * 4);
  float*  kg    = (float*)alloc((size_t)Bz * Nn * Tt * 4);
  float*  kgW   = (float*)alloc((size_t)Bz * Nn * Tt * 4);
  float*  ktb   = (float*)alloc((size_t)Bz * Tt * Nn * 4);
  __bf16* gW    = (__bf16*)alloc(O3 * Cc * 2);
  __bf16* rW    = (__bf16*)alloc(Oo * Cc * 2);
  __bf16* w1p   = (__bf16*)alloc(O3 * 2 * Cc * 2);
  __bf16* w2p   = (__bf16*)alloc(O3 * 2 * Cc * 2);

  k_pack<<<dim3((Oo * Cc + 255) / 256), dim3(256), 0, stream>>>(gacn_W, res_w, tcn1_w, tcn2_w, gW, rW, w1p, w2p);
  k_ln<<<dim3(Bz * Cc * Nn / 256), dim3(256), 0, stream>>>(x, ln_w, ln_b, h, hT, xb);
  k_kc<<<dim3(Bz * Cc * Tt / 256), dim3(256), 0, stream>>>(h, catt_al, kc);
  k_kgkt<<<dim3(Bz * Nn * Tt / 256), dim3(256), 0, stream>>>(h, gatt_al, tatt_al, kg, ktb);
  k_kW<<<dim3(Bz * Cc * Tt / 256), dim3(256), 0, stream>>>(kc, catt_W, kcW, Bz * Cc);
  k_kW<<<dim3(Bz * Nn * Tt / 256), dim3(256), 0, stream>>>(kg, gatt_W, kgW, Bz * Nn);
  k_att_c<<<dim3(Bz * Cc), dim3(64), 0, stream>>>(kcW, kc, attC);
  k_att_g<<<dim3(Bz * Nn), dim3(256), 0, stream>>>(kgW, kg, Adj, attG);
  k_att_t<<<dim3(Bz), dim3(256), 0, stream>>>(ktb, tatt_W1, tatt_W2, attTT);

  // c_out[b] = att_c[b] (64x64) @ h[b] (64x12288)
  k_gemm<false><<<dim3(NT / 64, 1, Bz), dim3(256), 0, stream>>>(
      attC, h, cOut, nullptr, 64, 64, NT,
      (long long)Cc * Cc, (long long)Cc * NT, (long long)Cc * NT, 0);
  // hW1[b] = gacn_W (64x64) @ h[b] (64x12288)
  k_gemm<false><<<dim3(NT / 64, 1, Bz), dim3(256), 0, stream>>>(
      gW, h, hW1, nullptr, 64, 64, NT,
      0LL, (long long)Cc * NT, (long long)Cc * NT, 0);
  k_transpose<<<dim3((unsigned)(SH / 256)), dim3(256), 0, stream>>>(hW1, hWT);
  // g_out[b] = att_g[b] (512x512) @ hWT[b] (512x1536)   (dominant GEMM, 12.9G MACs)
  k_gemm<false><<<dim3(O3 * Tt / 64, Nn / 64, Bz), dim3(256), 0, stream>>>(
      attG, hWT, hW1, nullptr, Nn, Nn, O3 * Tt,
      (long long)Nn * Nn, (long long)Nn * O3 * Tt, (long long)Nn * O3 * Tt, 0);
  // t_tmp[b] = h[b] (32768x24) @ att_t^T[b] (24x24)   (small, guarded path)
  k_gemm<true><<<dim3(1, Cc * Nn / 64, Bz), dim3(256), 0, stream>>>(
      h, attTT, tTmp, nullptr, Cc * Nn, Tt, Tt,
      (long long)Cc * Nn * Tt, (long long)Tt * Tt, (long long)Cc * Nn * Tt, 0);
  // two causal dilated convs (implicit GEMM, K=128)
  k_conv<<<dim3(NT / 64, 1, Bz), dim3(256), 0, stream>>>(w1p, tTmp, y1, tcn1_b, 1);
  k_conv<<<dim3(NT / 64, 1, Bz), dim3(256), 0, stream>>>(w2p, y1, tTmp, tcn2_b, 2);
  // residual: d_out[b] = res_w (192x64) @ x[b] (64x12288) + res_b  (f32 output)
  k_gemm<false><<<dim3(NT / 64, Oo / 64, Bz), dim3(256), 0, stream>>>(
      rW, xb, out, res_b, Oo, Cc, NT,
      0LL, (long long)Cc * NT, (long long)Oo * NT, 1);
  // relu(res + concat)
  k_final<<<dim3((unsigned)((size_t)Bz * Oo * NT / 256)), dim3(256), 0, stream>>>(
      out, cOut, hW1, tTmp);
}
